// HTP_7430293422821
// MI455X (gfx1250) — matrix operand order repaired and checked by reading, not compiled, not run on hardware
//
#include <hip/hip_runtime.h>
#include <hip/hip_bf16.h>

typedef __attribute__((ext_vector_type(16))) __bf16 v16bf;
typedef __attribute__((ext_vector_type(8)))  float  v8f;
typedef int v4i __attribute__((vector_size(16)));   // b128 payload type for async-LDS builtin
typedef __attribute__((address_space(1))) v4i* gp_b128;
typedef __attribute__((address_space(3))) v4i* lp_b128;

#define NXELE (2048 * 128)      // B*L*H
#define NEGV  (-4294967295.0f)  // -2^32 + 1

#if __has_builtin(__builtin_amdgcn_global_load_async_to_lds_b128) && \
    __has_builtin(__builtin_amdgcn_s_wait_asynccnt)
#define HAVE_ASYNC_LDS 1
#else
#define HAVE_ASYNC_LDS 0
#endif

__device__ __forceinline__ __bf16 f2bf(float f) {
  union { float f; unsigned u; } v; v.f = f;
  unsigned r = v.u + 0x7FFFu + ((v.u >> 16) & 1u);   // round-to-nearest-even
  unsigned short h = (unsigned short)(r >> 16);
  return __builtin_bit_cast(__bf16, h);
}
// contiguous k order k = 16*half + e, shared by A 16x32 and B 32x16 (16-bit)
__device__ __forceinline__ int ka_map(int half, int e) { return e + (half ? 16 : 0); }
__device__ __forceinline__ int kb_map(int half, int e) { return e + (half ? 16 : 0); }

// ---------------- LayerNorm: one wave per 128-wide row ----------------
__global__ void k_ln(const float* __restrict__ src, float* __restrict__ dst,
                     const float* __restrict__ g, const float* __restrict__ b) {
  int row = blockIdx.x, lane = threadIdx.x;
  const float* p = src + (size_t)row * 128;
  float4 v = *(const float4*)(p + lane * 4);
  float s = v.x + v.y + v.z + v.w;
  for (int d = 16; d >= 1; d >>= 1) s += __shfl_xor(s, d, 32);
  float mean = s * (1.0f / 128.0f);
  float dx = v.x - mean, dy = v.y - mean, dz = v.z - mean, dw = v.w - mean;
  float q = dx * dx + dy * dy + dz * dz + dw * dw;
  for (int d = 16; d >= 1; d >>= 1) q += __shfl_xor(q, d, 32);
  float inv = rsqrtf(q * (1.0f / 128.0f) + 1e-8f);
  int c = lane * 4;
  float4 o;
  o.x = dx * inv * g[c + 0] + b[c + 0];
  o.y = dy * inv * g[c + 1] + b[c + 1];
  o.z = dz * inv * g[c + 2] + b[c + 2];
  o.w = dw * inv * g[c + 3] + b[c + 3];
  *(float4*)(dst + (size_t)row * 128 + c) = o;
}

// ------------- C[M,128] = A[M,128] @ W[128,128]^T + bias (+relu/+res) -------------
// one wave per 16x64 tile: A fragment reused across 4 back-to-back WMMAs
template <int FLAGS>
__global__ void k_gemm_xWt(const float* __restrict__ A, const float* __restrict__ W,
                           const float* __restrict__ bias, const float* __restrict__ res,
                           float* __restrict__ C) {
  int lane = threadIdx.x, half = lane >> 4, l15 = lane & 15;
  int n0 = blockIdx.x * 64, m0 = blockIdx.y * 16;
  v8f acc[4] = {};
  for (int kk = 0; kk < 128; kk += 32) {
    v16bf af;
#pragma unroll
    for (int e = 0; e < 16; ++e)
      af[e] = f2bf(A[(size_t)(m0 + l15) * 128 + kk + ka_map(half, e)]);
#pragma unroll
    for (int t = 0; t < 4; ++t) {
      v16bf bf;
#pragma unroll
      for (int e = 0; e < 16; ++e)
        bf[e] = f2bf(W[(size_t)(n0 + t * 16 + l15) * 128 + kk + kb_map(half, e)]);  // B = W^T
      acc[t] = __builtin_amdgcn_wmma_f32_16x16x32_bf16(false, af, false, bf, (short)0, acc[t], false, false);
    }
  }
#pragma unroll
  for (int t = 0; t < 4; ++t)
#pragma unroll
    for (int r = 0; r < 8; ++r) {
      int m = m0 + r + (half ? 8 : 0), n = n0 + t * 16 + l15;
      float v = acc[t][r] + bias[n];
      if (FLAGS & 1) v = fmaxf(v, 0.0f);
      if (FLAGS & 2) v += res[(size_t)m * 128 + n];
      C[(size_t)m * 128 + n] = v;
    }
}

// ---- U_h[m,c] = sum_{k<32} Q[m, 32h+k] * WA1[32h+k, c]  (K=32: single WMMA step) ----
__global__ void k_gemm_U(const float* __restrict__ Q, const float* __restrict__ WA1,
                         float* __restrict__ U) {
  int h = blockIdx.z;
  int lane = threadIdx.x, half = lane >> 4, l15 = lane & 15;
  int n0 = blockIdx.x * 64, m0 = blockIdx.y * 16;
  v16bf af;
#pragma unroll
  for (int e = 0; e < 16; ++e)
    af[e] = f2bf(Q[(size_t)(m0 + l15) * 128 + 32 * h + ka_map(half, e)]);
  v8f acc[4] = {};
#pragma unroll
  for (int t = 0; t < 4; ++t) {
    v16bf bf;
#pragma unroll
    for (int e = 0; e < 16; ++e)
      bf[e] = f2bf(WA1[(size_t)(32 * h + kb_map(half, e)) * 128 + n0 + t * 16 + l15]);  // row-major B
    acc[t] = __builtin_amdgcn_wmma_f32_16x16x32_bf16(false, af, false, bf, (short)0, acc[t], false, false);
  }
#pragma unroll
  for (int t = 0; t < 4; ++t)
#pragma unroll
    for (int r = 0; r < 8; ++r)
      U[(size_t)h * NXELE + (size_t)(m0 + r + (half ? 8 : 0)) * 128 + n0 + t * 16 + l15] = acc[t][r];
}

// ---- pass A: S[b,h,i,j] = (Q_h.K_h + sum_c tm[b,i,j,c]*U_h[b,i,c]) * scale, masked ----
__global__ void k_scores(const float* __restrict__ Q, const float* __restrict__ Km,
                         const float* __restrict__ U, const float* __restrict__ tm,
                         const unsigned char* __restrict__ tmask, float* __restrict__ S) {
  int i = blockIdx.x, b = blockIdx.y;
  int tid = threadIdx.x, wave = tid >> 5, lane = tid & 31;
  int c0 = lane * 4, hc = lane >> 3;  // this lane's 4 channels live in head hc
  size_t rowq = (size_t)(b * 256 + i) * 128;
  const float* tmrow = tm + (size_t)(b * 256 + i) * 256 * 128;
  float4 qv = *(const float4*)(Q + rowq + c0);
  float4 uv[4];
#pragma unroll
  for (int hh = 0; hh < 4; ++hh)
    uv[hh] = *(const float4*)(U + (size_t)hh * NXELE + rowq + c0);
  bool rowmask = tmask[b * 256 + i] != 0;
  const float scale = 0.17677669529663687f;  // 1/sqrt(32)
  for (int j = wave * 32; j < wave * 32 + 32; ++j) {
    float4 t4 = *(const float4*)(tmrow + (size_t)j * 128 + c0);
    __builtin_prefetch(tmrow + (size_t)(j + 4) * 128 + c0, 0, 0);
    float4 kv = *(const float4*)(Km + (size_t)(b * 256 + j) * 128 + c0);
    float s[4];
#pragma unroll
    for (int hh = 0; hh < 4; ++hh)
      s[hh] = t4.x * uv[hh].x + t4.y * uv[hh].y + t4.z * uv[hh].z + t4.w * uv[hh].w;
    s[hc] += qv.x * kv.x + qv.y * kv.y + qv.z * kv.z + qv.w * kv.w;
#pragma unroll
    for (int hh = 0; hh < 4; ++hh)
      for (int d = 16; d >= 1; d >>= 1) s[hh] += __shfl_xor(s[hh], d, 32);
    if (lane < 4) {
      float val = s[lane] * scale;
      if (rowmask || j > i) val = NEGV;
      S[((size_t)(b * 4 + lane) * 256 + i) * 256 + j] = val;
    }
  }
}

// ---------------- softmax over last axis (rows of 256) ----------------
__global__ void k_softmax(float* __restrict__ S) {
  size_t row = blockIdx.x;
  float* p = S + row * 256;
  int tid = threadIdx.x, lane = tid & 31, wid = tid >> 5;
  __shared__ float redm[8], reds[8], bc[2];
  float v = p[tid];
  float m = v;
  for (int d = 16; d >= 1; d >>= 1) m = fmaxf(m, __shfl_xor(m, d, 32));
  if (lane == 0) redm[wid] = m;
  __syncthreads();
  if (tid == 0) {
    float mm = redm[0];
    for (int t = 1; t < 8; ++t) mm = fmaxf(mm, redm[t]);
    bc[0] = mm;
  }
  __syncthreads();
  float e = __expf(v - bc[0]);
  float s = e;
  for (int d = 16; d >= 1; d >>= 1) s += __shfl_xor(s, d, 32);
  if (lane == 0) reds[wid] = s;
  __syncthreads();
  if (tid == 0) {
    float ss = reds[0];
    for (int t = 1; t < 8; ++t) ss += reds[t];
    bc[1] = 1.0f / ss;
  }
  __syncthreads();
  p[tid] = e * bc[1];
}

// ---- pass B: T_h[b,i,c] = sum_j w[h,b,i,j] * tm[b,i,j,c] ----
// Async-LDS double-buffered: each wave pipelines 8-row (4KB) tm tiles through its
// private LDS staging buffer with global_load_async_to_lds_b128 / s_wait_asynccnt.
__global__ void k_timevalue(const float* __restrict__ Wp, const float* __restrict__ tm,
                            float* __restrict__ T) {
  int i = blockIdx.x, b = blockIdx.y;
  int tid = threadIdx.x, wave = tid >> 5, lane = tid & 31;
  __shared__ float acc[4 * 128];
  for (int t = tid; t < 512; t += 256) acc[t] = 0.0f;
  __syncthreads();
  const float* tmrow = tm + (size_t)(b * 256 + i) * 256 * 128;
  int c0 = lane * 4, j0 = wave * 32;
  float wreg[4];
#pragma unroll
  for (int hh = 0; hh < 4; ++hh)
    wreg[hh] = Wp[((size_t)(b * 4 + hh) * 256 + i) * 256 + j0 + lane];
  float a[4][4] = {};
#if HAVE_ASYNC_LDS
  __shared__ float stg[8][2][8 * 128];   // [wave][buf][8 rows x 128 ch] = 64KB
  float* buf[2] = { &stg[wave][0][0], &stg[wave][1][0] };
  // prime chunk 0
#pragma unroll
  for (int jj = 0; jj < 8; ++jj)
    __builtin_amdgcn_global_load_async_to_lds_b128(
        (gp_b128)(tmrow + (size_t)(j0 + jj) * 128 + c0),
        (lp_b128)(buf[0] + jj * 128 + c0), 0, 0);
  for (int ch = 0; ch < 4; ++ch) {
    if (ch < 3) {
#pragma unroll
      for (int jj = 0; jj < 8; ++jj)
        __builtin_amdgcn_global_load_async_to_lds_b128(
            (gp_b128)(tmrow + (size_t)(j0 + (ch + 1) * 8 + jj) * 128 + c0),
            (lp_b128)(buf[(ch + 1) & 1] + jj * 128 + c0), 0, 0);
      __builtin_amdgcn_s_wait_asynccnt(8);   // chunk ch landed; next chunk in flight
    } else {
      __builtin_amdgcn_s_wait_asynccnt(0);
    }
    const float* cb = buf[ch & 1];
#pragma unroll
    for (int jj = 0; jj < 8; ++jj) {
      float4 t4 = *(const float4*)(cb + jj * 128 + c0);
#pragma unroll
      for (int hh = 0; hh < 4; ++hh) {
        float wv = __shfl(wreg[hh], ch * 8 + jj, 32);
        a[hh][0] += wv * t4.x; a[hh][1] += wv * t4.y;
        a[hh][2] += wv * t4.z; a[hh][3] += wv * t4.w;
      }
    }
  }
#else
  for (int jj = 0; jj < 32; ++jj) {
    float4 t4 = *(const float4*)(tmrow + (size_t)(j0 + jj) * 128 + c0);
    __builtin_prefetch(tmrow + (size_t)(j0 + jj + 4) * 128 + c0, 0, 0);
#pragma unroll
    for (int hh = 0; hh < 4; ++hh) {
      float wv = __shfl(wreg[hh], jj, 32);
      a[hh][0] += wv * t4.x; a[hh][1] += wv * t4.y;
      a[hh][2] += wv * t4.z; a[hh][3] += wv * t4.w;
    }
  }
#endif
#pragma unroll
  for (int hh = 0; hh < 4; ++hh)
#pragma unroll
    for (int k = 0; k < 4; ++k) atomicAdd(&acc[hh * 128 + c0 + k], a[hh][k]);
  __syncthreads();
  for (int t = tid; t < 512; t += 256) {
    int hh = t >> 7, c = t & 127;
    T[(size_t)hh * NXELE + (size_t)(b * 256 + i) * 128 + c] = acc[t];
  }
}

// ---- AO[b,i,32h+dd] = sum_j P[h,b,i,j] * V[b,j,32h+dd]  (per-head w@V, 16x32/wave) ----
__global__ void k_wv(const float* __restrict__ P, const float* __restrict__ V,
                     float* __restrict__ AO) {
  int hb = blockIdx.y, h = hb >> 3, b = hb & 7;
  int lane = threadIdx.x, half = lane >> 4, l15 = lane & 15;
  int m0 = blockIdx.x * 16;
  const float* Pb = P + (size_t)(b * 4 + h) * 256 * 256;
  v8f acc[2] = {};
  for (int kk = 0; kk < 256; kk += 32) {
    v16bf af;
#pragma unroll
    for (int e = 0; e < 16; ++e)
      af[e] = f2bf(Pb[(size_t)(m0 + l15) * 256 + kk + ka_map(half, e)]);
#pragma unroll
    for (int t = 0; t < 2; ++t) {
      v16bf bf;
#pragma unroll
      for (int e = 0; e < 16; ++e)
        bf[e] = f2bf(V[(size_t)(b * 256 + kk + kb_map(half, e)) * 128 + 32 * h + t * 16 + l15]);
      acc[t] = __builtin_amdgcn_wmma_f32_16x16x32_bf16(false, af, false, bf, (short)0, acc[t], false, false);
    }
  }
#pragma unroll
  for (int t = 0; t < 2; ++t)
#pragma unroll
    for (int r = 0; r < 8; ++r)
      AO[(size_t)(b * 256 + m0 + r + (half ? 8 : 0)) * 128 + 32 * h + t * 16 + l15] = acc[t][r];
}

// ---- AO[m,32h+nn] += sum_c T_h[m,c] * WA2[32h+nn, c]  (16x32/wave, transposed-B) ----
__global__ void k_wa2(const float* __restrict__ T, const float* __restrict__ WA2,
                      float* __restrict__ AO) {
  int h = blockIdx.y;
  int lane = threadIdx.x, half = lane >> 4, l15 = lane & 15;
  int m0 = blockIdx.x * 16;
  const float* Th = T + (size_t)h * NXELE;
  v8f acc[2] = {};
  for (int kk = 0; kk < 128; kk += 32) {
    v16bf af;
#pragma unroll
    for (int e = 0; e < 16; ++e)
      af[e] = f2bf(Th[(size_t)(m0 + l15) * 128 + kk + ka_map(half, e)]);
#pragma unroll
    for (int t = 0; t < 2; ++t) {
      v16bf bf;
#pragma unroll
      for (int e = 0; e < 16; ++e)
        bf[e] = f2bf(WA2[(size_t)(32 * h + t * 16 + l15) * 128 + kk + kb_map(half, e)]);
      acc[t] = __builtin_amdgcn_wmma_f32_16x16x32_bf16(false, af, false, bf, (short)0, acc[t], false, false);
    }
  }
#pragma unroll
  for (int t = 0; t < 2; ++t)
#pragma unroll
    for (int r = 0; r < 8; ++r) {
      size_t idx = (size_t)(m0 + r + (half ? 8 : 0)) * 128 + 32 * h + t * 16 + l15;
      AO[idx] += acc[t][r];
    }
}

__global__ void k_add(float* __restrict__ dst, const float* __restrict__ a,
                      const float* __restrict__ b) {
  size_t i = (size_t)blockIdx.x * 256 + threadIdx.x;
  dst[i] = a[i] + b[i];
}

__global__ void k_keep(float* __restrict__ x, const unsigned char* __restrict__ tmask) {
  size_t i = (size_t)blockIdx.x * 256 + threadIdx.x;
  if (tmask[i >> 7]) x[i] = 0.0f;
}

extern "C" void kernel_launch(void* const* d_in, const int* in_sizes, int n_in,
                              void* d_out, int out_size, void* d_ws, size_t ws_size,
                              hipStream_t stream) {
  const float* seqs  = (const float*)d_in[0];
  const unsigned char* tmask = (const unsigned char*)d_in[1];
  const float* tm    = (const float*)d_in[2];
  const float* Qw = (const float*)d_in[3];  const float* Qb = (const float*)d_in[4];
  const float* Kw = (const float*)d_in[5];  const float* Kb = (const float*)d_in[6];
  const float* Vw = (const float*)d_in[7];  const float* Vb = (const float*)d_in[8];
  const float* WA1 = (const float*)d_in[9]; const float* WA2 = (const float*)d_in[10];
  const float* ln1g = (const float*)d_in[11]; const float* ln1b = (const float*)d_in[12];
  const float* ln2g = (const float*)d_in[13]; const float* ln2b = (const float*)d_in[14];
  const float* c1w = (const float*)d_in[15]; const float* c1b = (const float*)d_in[16];
  const float* c2w = (const float*)d_in[17]; const float* c2b = (const float*)d_in[18];
  const float* lnfg = (const float*)d_in[19]; const float* lnfb = (const float*)d_in[20];

  float* ws = (float*)d_ws;
  float* X  = ws;                      // [2048,128] activations
  float* Qn = ws + 1 * (size_t)NXELE;  // layernormed q
  float* Qm = ws + 2 * (size_t)NXELE;
  float* Km = ws + 3 * (size_t)NXELE;
  float* Vm = ws + 4 * (size_t)NXELE;
  float* U  = ws + 5 * (size_t)NXELE;  // 4 planes
  float* S  = ws + 9 * (size_t)NXELE;  // [8,4,256,256] scores/probs
  float* T  = ws + 17 * (size_t)NXELE; // 4 planes
  float* AO = ws + 21 * (size_t)NXELE;
  float* H1 = ws + 22 * (size_t)NXELE;

  (void)hipMemcpyAsync(X, seqs, (size_t)NXELE * sizeof(float), hipMemcpyDeviceToDevice, stream);

  for (int blk = 0; blk < 2; ++blk) {
    const float* Qw_ = Qw + (size_t)blk * 128 * 128; const float* Qb_ = Qb + blk * 128;
    const float* Kw_ = Kw + (size_t)blk * 128 * 128; const float* Kb_ = Kb + blk * 128;
    const float* Vw_ = Vw + (size_t)blk * 128 * 128; const float* Vb_ = Vb + blk * 128;
    const float* WA1_ = WA1 + (size_t)blk * 128 * 128;
    const float* WA2_ = WA2 + (size_t)blk * 128 * 128;
    const float* c1w_ = c1w + (size_t)blk * 128 * 128; const float* c1b_ = c1b + blk * 128;
    const float* c2w_ = c2w + (size_t)blk * 128 * 128; const float* c2b_ = c2b + blk * 128;

    k_ln<<<2048, 32, 0, stream>>>(X, Qn, ln1g + blk * 128, ln1b + blk * 128);
    k_gemm_xWt<0><<<dim3(2, 128), 32, 0, stream>>>(Qn, Qw_, Qb_, nullptr, Qm);
    k_gemm_xWt<0><<<dim3(2, 128), 32, 0, stream>>>(Qn, Kw_, Kb_, nullptr, Km);
    k_gemm_xWt<0><<<dim3(2, 128), 32, 0, stream>>>(Qn, Vw_, Vb_, nullptr, Vm);
    k_gemm_U<<<dim3(2, 128, 4), 32, 0, stream>>>(Qm, WA1_, U);
    k_scores<<<dim3(256, 8), 256, 0, stream>>>(Qm, Km, U, tm, tmask, S);
    k_softmax<<<8192, 256, 0, stream>>>(S);
    k_timevalue<<<dim3(256, 8), 256, 0, stream>>>(S, tm, T);
    k_wv<<<dim3(16, 32), 32, 0, stream>>>(S, Vm, AO);
    k_wa2<<<dim3(128, 4), 32, 0, stream>>>(T, WA2_, AO);
    k_add<<<1024, 256, 0, stream>>>(X, Qn, AO);          // x = q + attn
    k_ln<<<2048, 32, 0, stream>>>(X, X, ln2g + blk * 128, ln2b + blk * 128);
    k_gemm_xWt<1><<<dim3(2, 128), 32, 0, stream>>>(X, c1w_, c1b_, nullptr, H1);   // relu
    k_gemm_xWt<2><<<dim3(2, 128), 32, 0, stream>>>(H1, c2w_, c2b_, X, X);         // + residual
    k_keep<<<1024, 256, 0, stream>>>(X, tmask);
  }
  k_ln<<<2048, 32, 0, stream>>>(X, (float*)d_out, lnfg, lnfb);
}
